// StretcherDualSoftMaxMatcher_13761075216900
// MI455X (gfx1250) — matrix operand order repaired and checked by reading, not compiled, not run on hardware
//
#include <hip/hip_runtime.h>
#include <hip/hip_bf16.h>
#include <math.h>

// Problem constants (from reference setup_inputs)
#define T_DIM 27
#define N_DIM 4096
#define M_DIM 4096
#define D_DIM 256

typedef __attribute__((ext_vector_type(16))) __bf16 v16bf;
typedef __attribute__((ext_vector_type(8)))  __bf16 v8bf;
typedef __attribute__((ext_vector_type(8)))  float  v8f;

__device__ __forceinline__ unsigned short f2bf(float f) {
    unsigned int u = __float_as_uint(f);
    u += 0x7FFFu + ((u >> 16) & 1u);   // round-to-nearest-even
    return (unsigned short)(u >> 16);
}

// ---------------------------------------------------------------------------
// Kernel 1: L2-normalize each D=256 descriptor row, convert to bf16.
// One wave (32 lanes) per row; each lane handles 8 contiguous floats.
// Rows [0, T*N) come from stretched_descriptors, rows [T*N, T*N+M) from base.
// ---------------------------------------------------------------------------
__global__ __launch_bounds__(256)
void k_normalize(const float* __restrict__ sd, const float* __restrict__ base,
                 unsigned short* __restrict__ sdn, unsigned short* __restrict__ basen) {
    const int wid  = threadIdx.x >> 5;
    const int lane = threadIdx.x & 31;
    const long row = (long)blockIdx.x * 8 + wid;
    const long SDROWS = (long)T_DIM * N_DIM;

    const float* src;
    unsigned short* dst;
    if (row < SDROWS) {
        src = sd + row * D_DIM;
        dst = sdn + row * D_DIM;
    } else {
        long r = row - SDROWS;
        src = base + r * D_DIM;
        dst = basen + r * D_DIM;
    }

    const float4* s4 = (const float4*)(src + lane * 8);
    float4 a = s4[0];
    float4 b = s4[1];
    float ss = a.x*a.x + a.y*a.y + a.z*a.z + a.w*a.w
             + b.x*b.x + b.y*b.y + b.z*b.z + b.w*b.w;
    #pragma unroll
    for (int m = 16; m >= 1; m >>= 1)
        ss += __shfl_xor(ss, m, 32);
    const float sc = rsqrtf(ss);

    uint4 o;
    o.x = (unsigned)f2bf(a.x*sc) | ((unsigned)f2bf(a.y*sc) << 16);
    o.y = (unsigned)f2bf(a.z*sc) | ((unsigned)f2bf(a.w*sc) << 16);
    o.z = (unsigned)f2bf(b.x*sc) | ((unsigned)f2bf(b.y*sc) << 16);
    o.w = (unsigned)f2bf(b.z*sc) | ((unsigned)f2bf(b.w*sc) << 16);
    *(uint4*)(dst + lane * 8) = o;
}

// ---------------------------------------------------------------------------
// Kernel 2: WMMA GEMM with online max/argmax over T transforms.
// Block = 256 threads (8 waves), tile = 128(N) x 128(M).
// Wave (waveN in 0..3, waveM in 0..1) owns 32x64 -> 2x4 accumulators (v8f).
// A frag (16x32 bf16): lane 0-15 -> row l, K kb+0..7 & kb+16..23;
//                      lane 16-31 -> row l-16, K kb+8..15 & kb+24..31.
// B frag (32x16 bf16): lane 0-15 -> col l, K kb+0..7 & kb+16..23;
//                      lane 16-31 -> col l-16, K kb+8..15 & kb+24..31.
// Both patterns read contiguous 16B chunks along D (no transpose needed).
// ---------------------------------------------------------------------------
__global__ __launch_bounds__(256)
void k_gemm_max(const unsigned short* __restrict__ sdn_,
                const unsigned short* __restrict__ basen_,
                const int* __restrict__ itp,
                float* __restrict__ corr, float* __restrict__ idx_out) {
    const int lane  = threadIdx.x & 31;
    const int wid   = threadIdx.x >> 5;
    const int waveN = wid & 3;
    const int waveM = wid >> 2;
    const int rb = blockIdx.y * 128 + waveN * 32;   // N-dim base row for wave
    const int cb = blockIdx.x * 128 + waveM * 64;   // M-dim base col for wave
    const int l15 = lane & 15;
    const int lhi = lane >> 4;

    const __bf16* A = (const __bf16*)sdn_;
    const __bf16* B = (const __bf16*)basen_;

    const float itf = (float)(*itp);

    v8f best[2][4];
    v8f idxv[2][4];
    #pragma unroll
    for (int i = 0; i < 2; ++i)
        #pragma unroll
        for (int j = 0; j < 4; ++j)
            #pragma unroll
            for (int v = 0; v < 8; ++v) {
                best[i][j][v] = -INFINITY;
                idxv[i][j][v] = 0.0f;
            }

    for (int t = 0; t < T_DIM; ++t) {
        const __bf16* At = A + (size_t)t * N_DIM * D_DIM;

        v8f acc[2][4];
        #pragma unroll
        for (int i = 0; i < 2; ++i)
            #pragma unroll
            for (int j = 0; j < 4; ++j)
                acc[i][j] = (v8f){0.f, 0.f, 0.f, 0.f, 0.f, 0.f, 0.f, 0.f};

        #pragma unroll
        for (int kb = 0; kb < D_DIM; kb += 32) {
            v16bf af[2];
            v16bf bf[4];
            #pragma unroll
            for (int i = 0; i < 2; ++i) {
                const __bf16* p = At + (size_t)(rb + i * 16 + l15) * D_DIM + kb + lhi * 8;
                v8bf lo = *(const v8bf*)(p);
                v8bf hi = *(const v8bf*)(p + 16);
                af[i] = __builtin_shufflevector(lo, hi,
                        0,1,2,3,4,5,6,7,8,9,10,11,12,13,14,15);
            }
            #pragma unroll
            for (int j = 0; j < 4; ++j) {
                const __bf16* p = B + (size_t)(cb + j * 16 + l15) * D_DIM + kb + lhi * 8;
                v8bf lo = *(const v8bf*)(p);
                v8bf hi = *(const v8bf*)(p + 16);
                bf[j] = __builtin_shufflevector(lo, hi,
                        0,1,2,3,4,5,6,7,8,9,10,11,12,13,14,15);
            }
            #pragma unroll
            for (int i = 0; i < 2; ++i)
                #pragma unroll
                for (int j = 0; j < 4; ++j)
                    acc[i][j] = __builtin_amdgcn_wmma_f32_16x16x32_bf16(
                        false, af[i], false, bf[j], (short)0, acc[i][j], false, false);
        }

        const float ft = (float)t;
        #pragma unroll
        for (int i = 0; i < 2; ++i)
            #pragma unroll
            for (int j = 0; j < 4; ++j)
                #pragma unroll
                for (int v = 0; v < 8; ++v) {
                    float c = acc[i][j][v];
                    if (c > best[i][j][v]) {   // strict >, earliest t wins
                        best[i][j][v] = c;
                        idxv[i][j][v] = ft;
                    }
                }
    }

    // C/D layout: VGPR v -> row (v + 8*lhi), col l15 within the 16x16 tile.
    #pragma unroll
    for (int i = 0; i < 2; ++i)
        #pragma unroll
        for (int j = 0; j < 4; ++j) {
            const int m = cb + j * 16 + l15;
            #pragma unroll
            for (int v = 0; v < 8; ++v) {
                const int n = rb + i * 16 + v + lhi * 8;
                const size_t off = (size_t)n * M_DIM + m;
                corr[off]    = best[i][j][v] * itf;
                idx_out[off] = idxv[i][j][v];
            }
        }
}

// ---------------------------------------------------------------------------
// Kernel 3a: per-row (axis=1 over M) softmax stats: max + sum(exp).
// One block per row; 16 register-resident elements per thread.
// ---------------------------------------------------------------------------
__global__ __launch_bounds__(256)
void k_rowstats(const float* __restrict__ corr,
                float* __restrict__ rowmax, float* __restrict__ rowsum) {
    __shared__ float red[8];
    const int n = blockIdx.x;
    const float* r = corr + (size_t)n * M_DIM;
    const int lane = threadIdx.x & 31;
    const int wid  = threadIdx.x >> 5;

    float vals[16];
    #pragma unroll
    for (int k = 0; k < 16; ++k)
        vals[k] = r[threadIdx.x + k * 256];

    float mx = -INFINITY;
    #pragma unroll
    for (int k = 0; k < 16; ++k) mx = fmaxf(mx, vals[k]);
    #pragma unroll
    for (int s = 16; s >= 1; s >>= 1) mx = fmaxf(mx, __shfl_xor(mx, s, 32));
    if (lane == 0) red[wid] = mx;
    __syncthreads();
    mx = red[0];
    #pragma unroll
    for (int w = 1; w < 8; ++w) mx = fmaxf(mx, red[w]);
    __syncthreads();

    float s = 0.f;
    #pragma unroll
    for (int k = 0; k < 16; ++k) s += __expf(vals[k] - mx);
    #pragma unroll
    for (int m = 16; m >= 1; m >>= 1) s += __shfl_xor(s, m, 32);
    if (lane == 0) red[wid] = s;
    __syncthreads();
    if (threadIdx.x == 0) {
        float tot = 0.f;
        #pragma unroll
        for (int w = 0; w < 8; ++w) tot += red[w];
        rowmax[n] = mx;
        rowsum[n] = tot;
    }
}

// ---------------------------------------------------------------------------
// Kernel 3b: per-column (axis=0 over N) softmax stats. Thread-per-column,
// fully coalesced (consecutive threads read consecutive addresses).
// ---------------------------------------------------------------------------
__global__ __launch_bounds__(256)
void k_colstats(const float* __restrict__ corr,
                float* __restrict__ colmax, float* __restrict__ colsum) {
    const int m = blockIdx.x * 256 + threadIdx.x;
    float mx = -INFINITY;
    for (int n = 0; n < N_DIM; ++n)
        mx = fmaxf(mx, corr[(size_t)n * M_DIM + m]);
    float s = 0.f;
    for (int n = 0; n < N_DIM; ++n)
        s += __expf(corr[(size_t)n * M_DIM + m] - mx);
    colmax[m] = mx;
    colsum[m] = s;
}

// ---------------------------------------------------------------------------
// Kernel 4: P = softmax_N(corr) * softmax_M(corr), in place over corr.
// One block per row.
// ---------------------------------------------------------------------------
__global__ __launch_bounds__(256)
void k_dualsoftmax(float* __restrict__ corr,
                   const float* __restrict__ rowmax, const float* __restrict__ rowsum,
                   const float* __restrict__ colmax, const float* __restrict__ colsum) {
    const int n = blockIdx.x;
    const float rm = rowmax[n];
    const float inv_rs = 1.0f / rowsum[n];
    float* r = corr + (size_t)n * M_DIM;
    #pragma unroll
    for (int k = 0; k < 16; ++k) {
        const int m = threadIdx.x + k * 256;
        const float c = r[m];
        const float p = __expf(c - rm) * inv_rs * (__expf(c - colmax[m]) / colsum[m]);
        r[m] = p;
    }
}

// ---------------------------------------------------------------------------
extern "C" void kernel_launch(void* const* d_in, const int* in_sizes, int n_in,
                              void* d_out, int out_size, void* d_ws, size_t ws_size,
                              hipStream_t stream) {
    (void)in_sizes; (void)n_in; (void)out_size; (void)ws_size;

    const float* sd   = (const float*)d_in[0];   // [T, N, D] f32
    const float* base = (const float*)d_in[1];   // [1, M, D] f32
    const int*   itp  = (const int*)d_in[2];     // scalar inv_temp

    float* out  = (float*)d_out;
    float* corr = out;                                   // [N, M] -> becomes P in place
    float* idx  = out + (size_t)N_DIM * M_DIM;           // [N, M] float indices

    // Workspace layout: bf16 normalized sd | bf16 normalized base | stats
    unsigned short* sdn   = (unsigned short*)d_ws;
    unsigned short* basen = sdn + (size_t)T_DIM * N_DIM * D_DIM;
    float* rowmax = (float*)(basen + (size_t)M_DIM * D_DIM);
    float* rowsum = rowmax + N_DIM;
    float* colmax = rowsum + N_DIM;
    float* colsum = colmax + M_DIM;

    const int total_rows = T_DIM * N_DIM + M_DIM;        // 114688
    k_normalize<<<total_rows / 8, 256, 0, stream>>>(sd, base, sdn, basen);

    dim3 g2(M_DIM / 128, N_DIM / 128);                   // 32 x 32 tiles
    k_gemm_max<<<g2, 256, 0, stream>>>(sdn, basen, itp, corr, idx);

    k_rowstats<<<N_DIM, 256, 0, stream>>>(corr, rowmax, rowsum);
    k_colstats<<<M_DIM / 256, 256, 0, stream>>>(corr, colmax, colsum);
    k_dualsoftmax<<<N_DIM, 256, 0, stream>>>(corr, rowmax, rowsum, colmax, colsum);
}
